// LossLinkerE2EHoi_24790551232622
// MI455X (gfx1250) — matrix operand order repaired and checked by reading, not compiled, not run on hardware
//
#include <hip/hip_runtime.h>
#include <hip/hip_bf16.h>
#include <stdint.h>

// Problem constants (from reference)
#define B_     2
#define N_     4096
#define K_     2048
#define C_     16
#define DSPAN_ 2048
#define E_     300
#define H_     512

typedef __attribute__((ext_vector_type(16))) __bf16 v16bf;
typedef __attribute__((ext_vector_type(8)))  __bf16 v8bf;
typedef __attribute__((ext_vector_type(4)))  __bf16 v4bf;
typedef __attribute__((ext_vector_type(8)))  float  v8f;

// Packed B-fragment geometry (wave32 WMMA bf16 B layout: lane = N column, elems = K)
#define NTILES       (H_ / 16)       // 32 column tiles of W1
#define KSTEPS_SPAN  (DSPAN_ / 32)   // 64
#define KSTEPS_ENT   10              // ceil(300/32), zero padded to 320
#define FRAG_ELEMS   512             // 32 lanes * 16 bf16
#define W1S_ELEMS    (NTILES * KSTEPS_SPAN * FRAG_ELEMS)   // 1,048,576
#define W1E_ELEMS    (NTILES * KSTEPS_ENT  * FRAG_ELEMS)   //   163,840

// Workspace layout (bytes)
#define OFF_W1S      ((size_t)0)
#define OFF_W1E      ((size_t)(W1S_ELEMS * 2))                         // 2,097,152
#define OFF_SPANPART (OFF_W1E + (size_t)(W1E_ELEMS * 2))               // 2,424,832
#define OFF_PARTIALS (OFF_SPANPART + (size_t)B_ * K_ * H_ * 4)         // 10,813,440
// total workspace need ~10.83 MB

__device__ __forceinline__ __bf16 cvt_bf16(float f) {
    return static_cast<__bf16>(f);                 // RNE; packs to v_cvt_pk_bf16_f32
}
__device__ __forceinline__ unsigned short cvt_bf16_bits(float f) {
    return __builtin_bit_cast(unsigned short, static_cast<__bf16>(f));
}
__device__ __forceinline__ v8f vzero8() {
    v8f z = {0.f, 0.f, 0.f, 0.f, 0.f, 0.f, 0.f, 0.f};
    return z;
}

// ---------------------------------------------------------------------------
// Kernel 1: pack W1 (f32, [2348 x 512] row-major) into bf16 WMMA B-fragments.
// frag[(ntile*KSTEPS + ks)*512 + lane*16 + e]:
//   column h = ntile*16 + (lane&15);  row d = ks*32 + (e&7) + 8*(lane>>4) + 16*(e>>3)
// Entity part uses rows 2048..2347 of W1, zero-padded to K=320.
// ---------------------------------------------------------------------------
__global__ __launch_bounds__(256) void pack_w1_kernel(
    const float* __restrict__ W1,
    unsigned short* __restrict__ w1s,
    unsigned short* __restrict__ w1e)
{
    int i = blockIdx.x * 256 + threadIdx.x;
    if (i < W1S_ELEMS) {
        int ntile = i / (KSTEPS_SPAN * FRAG_ELEMS);
        int rem   = i % (KSTEPS_SPAN * FRAG_ELEMS);
        int ks    = rem / FRAG_ELEMS;
        int q     = rem % FRAG_ELEMS;
        int lane  = q >> 4, e = q & 15;
        int d     = ks * 32 + (e & 7) + 8 * (lane >> 4) + 16 * (e >> 3);
        int h     = ntile * 16 + (lane & 15);
        w1s[i] = cvt_bf16_bits(W1[(size_t)d * H_ + h]);
    } else if (i < W1S_ELEMS + W1E_ELEMS) {
        int j     = i - W1S_ELEMS;
        int ntile = j / (KSTEPS_ENT * FRAG_ELEMS);
        int rem   = j % (KSTEPS_ENT * FRAG_ELEMS);
        int ks    = rem / FRAG_ELEMS;
        int q     = rem % FRAG_ELEMS;
        int lane  = q >> 4, e = q & 15;
        int d     = ks * 32 + (e & 7) + 8 * (lane >> 4) + 16 * (e >> 3);
        int h     = ntile * 16 + (lane & 15);
        w1e[j] = (d < E_) ? cvt_bf16_bits(W1[(size_t)(DSPAN_ + d) * H_ + h])
                          : (unsigned short)0;
    }
}

// ---------------------------------------------------------------------------
// Kernel 2: span GEMM.  span_part[B*K, 512] = gathered_span[B*K, 2048] @ W1_span
// 32 rows (2 M-tiles) per block to halve W1 re-reads; A tile staged to LDS in
// WMMA fragment layout (bf16, converted once per block), double-buffered.
// Per k-step each wave: 4 B-frag global loads, 2 A ds-loads, 8 WMMAs.
// A layout (16-bit A 16x32): lane m=lane&15, half=lane>>4;
//   elems 0..7 -> K = kb + 8*half + e; elems 8..15 -> K = kb + 16 + 8*half + (e-8)
// ---------------------------------------------------------------------------
__global__ __launch_bounds__(256) void span_gemm_kernel(
    const float* __restrict__ csv,          // [B,N,DSPAN]
    const int*   __restrict__ lidx,         // [B,K]
    const unsigned short* __restrict__ w1s_u16,
    float* __restrict__ span_part)          // [B*K, H]
{
    __shared__ alignas(32) unsigned short sA[2][2 * FRAG_ELEMS];   // 4 KB

    const v16bf* bfr = reinterpret_cast<const v16bf*>(w1s_u16);
    const int tid  = threadIdx.x;
    const int w    = tid >> 5, lane = tid & 31;
    const int half = lane >> 4, n = lane & 15;

    // --- staging role: this thread owns 4 consecutive fragment elements ---
    const int sq    = tid * 4;          // element index in [0, 1024)
    const int smt   = sq >> 9;          // which of the 2 M-tiles
    const int sqq   = sq & 511;         // element within tile fragment
    const int slane = sqq >> 4;
    const int se0   = sqq & 15;         // 0,4,8,12 (stays within one K chunk)
    const int sm    = slane & 15, shalf = slane >> 4;

    const int  sgm  = blockIdx.x * 32 + smt * 16 + sm;
    const int  sbb  = sgm >> 11;
    const int  skk  = sgm & (K_ - 1);
    const int  sidx = lidx[sbb * K_ + skk];
    const float* sarow = csv + ((size_t)sbb * N_ + (size_t)sidx) * DSPAN_;
    const int  skoff = (se0 < 8) ? (8 * shalf + se0) : (16 + 8 * shalf + (se0 - 8));

    auto stage = [&](int ks, int par) {
        const float4 f = *(const float4*)(sarow + ks * 32 + skoff);
        v4bf p;
        p[0] = cvt_bf16(f.x); p[1] = cvt_bf16(f.y);
        p[2] = cvt_bf16(f.z); p[3] = cvt_bf16(f.w);
        *(v4bf*)(&sA[par][smt * FRAG_ELEMS + sqq]) = p;
    };

    v8f acc[8];
    #pragma unroll
    for (int i = 0; i < 8; ++i) acc[i] = vzero8();

    stage(0, 0);
    for (int ks = 0; ks < KSTEPS_SPAN; ++ks) {
        __syncthreads();                       // staged buffer (ks&1) ready
        if (ks + 1 < KSTEPS_SPAN) stage(ks + 1, (ks + 1) & 1);

        v16bf bfrag[4];
        #pragma unroll
        for (int nt = 0; nt < 4; ++nt)
            bfrag[nt] = bfr[((size_t)(w * 4 + nt) * KSTEPS_SPAN + ks) * 32 + lane];

        const int par = ks & 1;
        #pragma unroll
        for (int mt = 0; mt < 2; ++mt) {
            v16bf a = *(const v16bf*)(&sA[par][mt * FRAG_ELEMS + lane * 16]);
            #pragma unroll
            for (int nt = 0; nt < 4; ++nt)
                acc[mt * 4 + nt] = __builtin_amdgcn_wmma_f32_16x16x32_bf16(
                    false, a, false, bfrag[nt], (short)0, acc[mt * 4 + nt],
                    false, false);
        }
    }

    // D layout: elem r -> (M = r + 8*half, N = lane&15)
    #pragma unroll
    for (int mt = 0; mt < 2; ++mt) {
        #pragma unroll
        for (int nt = 0; nt < 4; ++nt) {
            const int h = (w * 4 + nt) * 16 + n;
            #pragma unroll
            for (int r = 0; r < 8; ++r) {
                const int row = blockIdx.x * 32 + mt * 16 + r + 8 * half;
                span_part[(size_t)row * H_ + h] = acc[mt * 4 + nt][r];
            }
        }
    }
}

// ---------------------------------------------------------------------------
// Kernel 3: fused entity GEMM + epilogue. 4 (b,k) groups per block (amortizes
// W1_ent fragment reads 4x). A = 16 candidate embeddings per group, staged to
// LDS (bf16 fragment layout, converted once), double-buffered.
// Per k-step each wave: 4 B-frag loads reused over 4 groups -> 16 WMMAs.
// Epilogue: h = relu(ent + span_part + b1); score = h.W2 + b2; masked BCE;
// deterministic LDS reductions -> per-group partial.
// ---------------------------------------------------------------------------
__global__ __launch_bounds__(256) void ent_fused_kernel(
    const float* __restrict__ emb,     // [V,E]
    const float* __restrict__ tg,      // [B,N,C]
    const float* __restrict__ b1,      // [H]
    const float* __restrict__ W2,      // [H]
    const float* __restrict__ b2,      // [1]
    const int*   __restrict__ lidx,    // [B,K]
    const int*   __restrict__ candA,   // [B,N,C]
    const int*   __restrict__ clens,   // [B,N]
    const unsigned short* __restrict__ w1e_u16,
    const float* __restrict__ span_part,  // [B*K, H]
    float* __restrict__ partials)         // [B*K]
{
    __shared__ alignas(32) unsigned short sA[2][4 * FRAG_ELEMS];  // 8 KB
    __shared__ int   s_idx[4];
    __shared__ int   s_cand[64];
    __shared__ float s_part[4 * 8 * 32 * 8];                      // 32 KB
    __shared__ float s_red[64];

    const int g0  = blockIdx.x * 4;
    const int tid = threadIdx.x;
    const int w   = tid >> 5, lane = tid & 31;
    const int half = lane >> 4, n = lane & 15;

    if (tid < 4) s_idx[tid] = lidx[g0 + tid];
    __syncthreads();
    if (tid < 64) {
        const int gi = tid >> 4;
        const int bb = (g0 + gi) >> 11;
        s_cand[tid] = candA[((size_t)bb * N_ + (size_t)s_idx[gi]) * C_ + (tid & 15)];
    }
    __syncthreads();

    // --- staging role: this thread owns 8 consecutive fragment elements ---
    const int sq    = tid * 8;          // element index in [0, 2048)
    const int sgi   = sq >> 9;          // group
    const int sqq   = sq & 511;
    const int slane = sqq >> 4;
    const int se0   = sqq & 15;         // 0 or 8 (one full K chunk)
    const int sm    = slane & 15, shalf = slane >> 4;
    const int seid  = s_cand[sgi * 16 + sm];
    const float* serow = emb + (size_t)seid * E_;
    const int  skoff = (se0 == 0) ? (8 * shalf) : (16 + 8 * shalf);

    auto stage = [&](int ks, int par) {
        const int kb = ks * 32 + skoff;
        v8bf p;
        #pragma unroll
        for (int j = 0; j < 8; ++j) {
            const int kq = kb + j;
            p[j] = cvt_bf16(kq < E_ ? serow[kq] : 0.f);
        }
        *(v8bf*)(&sA[par][sgi * FRAG_ELEMS + sqq]) = p;
    };

    const v16bf* bfr = reinterpret_cast<const v16bf*>(w1e_u16);

    v8f acc[16];
    #pragma unroll
    for (int i = 0; i < 16; ++i) acc[i] = vzero8();

    stage(0, 0);
    for (int ks = 0; ks < KSTEPS_ENT; ++ks) {
        __syncthreads();
        if (ks + 1 < KSTEPS_ENT) stage(ks + 1, (ks + 1) & 1);

        v16bf bfrag[4];
        #pragma unroll
        for (int nt = 0; nt < 4; ++nt)
            bfrag[nt] = bfr[((size_t)(w * 4 + nt) * KSTEPS_ENT + ks) * 32 + lane];

        const int par = ks & 1;
        #pragma unroll
        for (int gi = 0; gi < 4; ++gi) {
            v16bf a = *(const v16bf*)(&sA[par][gi * FRAG_ELEMS + lane * 16]);
            #pragma unroll
            for (int nt = 0; nt < 4; ++nt)
                acc[gi * 4 + nt] = __builtin_amdgcn_wmma_f32_16x16x32_bf16(
                    false, a, false, bfrag[nt], (short)0, acc[gi * 4 + nt],
                    false, false);
        }
    }

    // Epilogue: per-lane column partial score contributions, per group.
    #pragma unroll
    for (int gi = 0; gi < 4; ++gi) {
        float partial[8];
        #pragma unroll
        for (int r = 0; r < 8; ++r) partial[r] = 0.f;

        #pragma unroll
        for (int nt = 0; nt < 4; ++nt) {
            const int h = (w * 4 + nt) * 16 + n;
            const float sp   = span_part[(size_t)(g0 + gi) * H_ + h];
            const float bias = b1[h];
            const float w2v  = W2[h];
            #pragma unroll
            for (int r = 0; r < 8; ++r) {
                float hv = acc[gi * 4 + nt][r] + sp + bias; // candidate c=r+8*half
                hv = fmaxf(hv, 0.f);                        // ReLU
                partial[r] = __builtin_fmaf(w2v, hv, partial[r]);
            }
        }
        #pragma unroll
        for (int r = 0; r < 8; ++r)
            s_part[((gi * 8 + w) * 32 + lane) * 8 + r] = partial[r];
    }
    __syncthreads();

    // Deterministic cross-wave/lane reduction per (group, candidate).
    if (tid < 64) {
        const int gi = tid >> 4, c = tid & 15;
        const int r = c & 7, hn = c >> 3;
        float sum = 0.f;
        for (int ww = 0; ww < 8; ++ww)
            for (int l = 16 * hn; l < 16 * hn + 16; ++l)
                sum += s_part[((gi * 8 + ww) * 32 + l) * 8 + r];

        const int bb  = (g0 + gi) >> 11;
        const int idx = s_idx[gi];
        const float score = sum + b2[0];
        const float t = tg[((size_t)bb * N_ + (size_t)idx) * C_ + c];
        const int clen = clens[bb * N_ + idx];
        // stable softplus(x) - x*t
        const float bce = fmaxf(score, 0.f) - score * t
                        + log1pf(expf(-fabsf(score)));
        s_red[tid] = (c < clen) ? bce : 0.f;
    }
    __syncthreads();

    if (tid < 4) {
        float s = 0.f;
        for (int c = 0; c < 16; ++c) s += s_red[tid * 16 + c];
        partials[g0 + tid] = s;
    }
}

// ---------------------------------------------------------------------------
// Kernel 4: deterministic final reduction of 4096 block partials -> scalar.
// ---------------------------------------------------------------------------
__global__ __launch_bounds__(256) void reduce_kernel(
    const float* __restrict__ partials, float* __restrict__ out)
{
    __shared__ float s[256];
    const int tid = threadIdx.x;
    float v = 0.f;
    for (int i = 0; i < (B_ * K_) / 256; ++i)
        v += partials[tid + 256 * i];
    s[tid] = v;
    __syncthreads();
    for (int off = 128; off > 0; off >>= 1) {
        if (tid < off) s[tid] += s[tid + off];
        __syncthreads();
    }
    if (tid == 0) out[0] = s[0] * 1.0f;   // WEIGHT = 1.0
}

// ---------------------------------------------------------------------------
extern "C" void kernel_launch(void* const* d_in, const int* in_sizes, int n_in,
                              void* d_out, int out_size, void* d_ws, size_t ws_size,
                              hipStream_t stream)
{
    (void)in_sizes; (void)n_in; (void)out_size; (void)ws_size;

    const float* csv   = (const float*)d_in[0];  // cand_span_vecs [B,N,DSPAN]
    const float* tg    = (const float*)d_in[1];  // targets_all    [B,N,C]
    const float* emb   = (const float*)d_in[2];  // entity_emb     [V,E]
    const float* W1    = (const float*)d_in[3];  // [DSPAN+E, H]
    const float* b1    = (const float*)d_in[4];  // [H]
    const float* W2    = (const float*)d_in[5];  // [H,1]
    const float* b2    = (const float*)d_in[6];  // [1]
    const int*   lidx  = (const int*)d_in[7];    // linker_indices_hoi [B,K]
    const int*   candA = (const int*)d_in[8];    // candidates_all     [B,N,C]
    const int*   clens = (const int*)d_in[9];    // candidate_lengths  [B,N]

    char* ws = (char*)d_ws;
    unsigned short* w1s      = (unsigned short*)(ws + OFF_W1S);
    unsigned short* w1e      = (unsigned short*)(ws + OFF_W1E);
    float*          spanpart = (float*)(ws + OFF_SPANPART);
    float*          partials = (float*)(ws + OFF_PARTIALS);
    float*          out      = (float*)d_out;

    const int packTotal = W1S_ELEMS + W1E_ELEMS;
    pack_w1_kernel<<<(packTotal + 255) / 256, 256, 0, stream>>>(W1, w1s, w1e);
    span_gemm_kernel<<<(B_ * K_) / 32, 256, 0, stream>>>(csv, lidx, w1s, spanpart);
    ent_fused_kernel<<<(B_ * K_) / 4, 256, 0, stream>>>(emb, tg, b1, W2, b2,
                                                        lidx, candA, clens,
                                                        w1e, spanpart, partials);
    reduce_kernel<<<1, 256, 0, stream>>>(partials, out);
}
